// OptimizedSelfAttention_32701880991970
// MI455X (gfx1250) — compile-verified, matrix operand order checked by reading, not run on hardware
//
#include <hip/hip_runtime.h>

#define HID   2048
#define NHEADS 16
#define DK    128
#define TSEQ  2048
#define NB    4
#define QKV_N 6144
#define MTOT  (NB*TSEQ)   // 8192

typedef __attribute__((ext_vector_type(16))) __bf16 v16bf;
typedef __attribute__((ext_vector_type(8)))  float  v8f;
typedef __attribute__((ext_vector_type(4)))  unsigned int v4u;
typedef __attribute__((ext_vector_type(8)))  int  v8i;
typedef __attribute__((ext_vector_type(4)))  int  v4i;

union Frag16 { v16bf v; unsigned int u[8]; unsigned short h[16]; };

#if defined(__has_builtin)
# if __has_builtin(__builtin_amdgcn_tensor_load_to_lds) && __has_builtin(__builtin_amdgcn_s_wait_tensorcnt)
#  define USE_TDM 1
# endif
#endif
#ifndef USE_TDM
# define USE_TDM 0
#endif

__device__ __forceinline__ unsigned short f2bf(float f) {
  unsigned int x = __float_as_uint(f);
  return (unsigned short)((x + 0x7fffu + ((x >> 16) & 1u)) >> 16);
}
__device__ __forceinline__ unsigned int pk2bf(float lo, float hi) {
  return (unsigned int)f2bf(lo) | ((unsigned int)f2bf(hi) << 16);
}
// load a 16x32 bf16 WMMA operand fragment: two contiguous 16B chunks
__device__ __forceinline__ void ldsFrag(Frag16& f, const unsigned short* p) {
  *(uint4*)&f.u[0] = *(const uint4*)(p);
  *(uint4*)&f.u[4] = *(const uint4*)(p + 16);
}

// ---------------- Kernel 1: QKV GEMM  (hidden @ w_qkv^T) ----------------
__global__ __launch_bounds__(256) void qkv_gemm_kernel(
    const float* __restrict__ A, const float* __restrict__ W,
    unsigned short* __restrict__ Qb, unsigned short* __restrict__ Kb,
    unsigned short* __restrict__ Vb)
{
  __shared__ unsigned short sA[128*40];
  __shared__ unsigned short sB[128*40];
  const int tid = threadIdx.x;
  const int w = tid >> 5, l = tid & 31, ln = l & 15, lh = l >> 4;
  const int m0 = blockIdx.y * 128, n0 = blockIdx.x * 128;

  v8f acc[8];
  for (int i = 0; i < 8; i++) for (int j = 0; j < 8; j++) acc[i][j] = 0.0f;

  for (int k0 = 0; k0 < HID; k0 += 32) {
    // fp32 -> bf16 conversion fused into LDS staging (float4 loads, b64 LDS stores)
    for (int idx = tid; idx < 128*8; idx += 256) {
      int r = idx >> 3, c = (idx & 7) * 4;
      float4 a4 = *(const float4*)&A[(size_t)(m0 + r)*HID + k0 + c];
      float4 w4 = *(const float4*)&W[(size_t)(n0 + r)*HID + k0 + c];
      unsigned long long pa = (unsigned long long)pk2bf(a4.x, a4.y)
                            | ((unsigned long long)pk2bf(a4.z, a4.w) << 32);
      unsigned long long pb = (unsigned long long)pk2bf(w4.x, w4.y)
                            | ((unsigned long long)pk2bf(w4.z, w4.w) << 32);
      *(unsigned long long*)&sA[r*40 + c] = pa;
      *(unsigned long long*)&sB[r*40 + c] = pb;
    }
    if (k0 + 32 < HID) {
      __builtin_prefetch(&A[(size_t)(m0 + (tid >> 1))*HID + k0 + 32], 0, 1);
      __builtin_prefetch(&W[(size_t)(n0 + (tid >> 1))*HID + k0 + 32], 0, 1);
    }
    __syncthreads();

    Frag16 fa;
    ldsFrag(fa, &sA[(16*w + ln)*40 + 8*lh]);
    #pragma unroll
    for (int nc = 0; nc < 8; nc++) {
      Frag16 fb;
      ldsFrag(fb, &sB[(nc*16 + ln)*40 + 8*lh]);
      acc[nc] = __builtin_amdgcn_wmma_f32_16x16x32_bf16(
          false, fa.v, false, fb.v, (short)0, acc[nc], false, false);
    }
    __syncthreads();
  }

  const float qscale = 0.08838834764831845f; // 1/sqrt(128)
  #pragma unroll
  for (int nc = 0; nc < 8; nc++) {
    #pragma unroll
    for (int r = 0; r < 8; r++) {
      int gm = m0 + 16*w + r + 8*lh;
      int gn = n0 + nc*16 + ln;
      int b = gm >> 11, t = gm & (TSEQ - 1);
      float v = acc[nc][r];
      if (gn < 2048) {
        int h = gn >> 7, d = gn & 127;
        Qb[(((size_t)(b*NHEADS + h))*TSEQ + t)*DK + d] = f2bf(v * qscale);
      } else if (gn < 4096) {
        int c = gn - 2048, h = c >> 7, d = c & 127;
        Kb[(((size_t)(b*NHEADS + h))*TSEQ + t)*DK + d] = f2bf(v);
      } else {
        int c = gn - 4096, h = c >> 7, d = c & 127;
        Vb[(((size_t)(b*NHEADS + h))*TSEQ + t)*DK + d] = f2bf(v);
      }
    }
  }
}

// ---------------- Kernel 2: causal flash attention ----------------
__global__ __launch_bounds__(256) void flash_attn_kernel(
    const unsigned short* __restrict__ Qb, const unsigned short* __restrict__ Kb,
    const unsigned short* __restrict__ Vb, unsigned short* __restrict__ AO)
{
  __shared__ unsigned short sK[32*136];    // K tile [s][k], 272B row stride
  __shared__ unsigned short sVt[128*40];   // V tile transposed [d][s]
  __shared__ unsigned short sP[8*16*40];   // per-wave P scratch

  const int tid = threadIdx.x;
  const int w = tid >> 5, l = tid & 31, ln = l & 15, lh = l >> 4;
  const int bh = blockIdx.x >> 4;
  const int q0 = (blockIdx.x & 15) * 128;
  const int b = bh >> 4, h = bh & 15;
  const int wq0 = q0 + 16*w;

  const unsigned int* V32 = (const unsigned int*)Vb;

  // Q fragments (16 rows x 128 k): 8 global b128 loads, kept in registers
  unsigned int qf[4][8];
  {
    const uint4* Qv = (const uint4*)Qb;
    const size_t rb = ((size_t)bh*TSEQ + wq0 + ln) * (DK/8);
    #pragma unroll
    for (int kc = 0; kc < 4; kc++) {
      uint4 a = Qv[rb + kc*4 + lh];
      uint4 c = Qv[rb + kc*4 + 2 + lh];
      qf[kc][0]=a.x; qf[kc][1]=a.y; qf[kc][2]=a.z; qf[kc][3]=a.w;
      qf[kc][4]=c.x; qf[kc][5]=c.y; qf[kc][6]=c.z; qf[kc][7]=c.w;
    }
  }

  v8f oacc[8];
  for (int i = 0; i < 8; i++) for (int j = 0; j < 8; j++) oacc[i][j] = 0.f;
  float Mr[8], Lr[8];
  #pragma unroll
  for (int r = 0; r < 8; r++) { Mr[r] = -3.0e38f; Lr[r] = 0.f; }

  const int pbase = w * (16*40);

  for (int s0 = 0; s0 < q0 + 128; s0 += 32) {
#if USE_TDM
    // Tensor Data Mover: 2D tile (64 dwords x 32 rows) global->LDS with
    // 4-dword pad after every 64 dwords -> 272B LDS row stride.
    if (w == 0) {
      unsigned long long ga = (unsigned long long)(size_t)Kb
                            + ((unsigned long long)bh*TSEQ + (unsigned)s0) * (DK*2);
      unsigned int lds = (unsigned int)(size_t)&sK[0];
      v4u g0;
      g0[0] = 1u;                                  // count=1, user mode
      g0[1] = lds;                                 // lds_addr
      g0[2] = (unsigned int)(ga & 0xffffffffu);    // global_addr[31:0]
      g0[3] = (unsigned int)((ga >> 32) & 0x1ffffffu) | (2u << 30); // [56:32], type=2
      v8i g1;
      g1[0] = (int)((2u << 16) | (1u << 20) | (5u << 22) | (3u << 25)); // 4B, pad 64+4
      g1[1] = (int)(64u << 16);                    // tensor_dim0 = 64 dwords
      g1[2] = (int)(32u << 16);                    // tensor_dim1 = 32 rows
      g1[3] = (int)(64u << 16);                    // tile_dim0 = 64
      g1[4] = 32;                                  // tile_dim1 = 32
      g1[5] = 64;                                  // tensor_dim0_stride = 64 dwords
      g1[6] = 0; g1[7] = 0;
      v4i gz = {0, 0, 0, 0};
      v8i gz8 = {0, 0, 0, 0, 0, 0, 0, 0};
      __builtin_amdgcn_tensor_load_to_lds(g0, g1, gz, gz, gz8, 0);
    }
#else
    for (int idx = tid; idx < 32*16; idx += 256) {   // b128 copy fallback
      int r = idx >> 4, c8 = (idx & 15) * 8;         // halves
      uint4 kv = *(const uint4*)&Kb[((size_t)bh*TSEQ + s0 + r)*DK + c8];
      *(uint4*)&sK[r*136 + c8] = kv;
    }
#endif
    // V tile transposed (all threads)
    for (int idx = tid; idx < 32*64; idx += 256) {
      int r = idx >> 6, c2 = idx & 63;
      unsigned int vv = V32[((size_t)bh*TSEQ + s0 + r)*(DK/2) + c2];
      sVt[(2*c2    )*40 + r] = (unsigned short)(vv & 0xffffu);
      sVt[(2*c2 + 1)*40 + r] = (unsigned short)(vv >> 16);
    }
#if USE_TDM
    if (w == 0) __builtin_amdgcn_s_wait_tensorcnt(0);
#endif
    __syncthreads();

    if (s0 <= wq0 + 15) {   // wave-uniform causal skip
      v8f sacc[2];
      for (int i = 0; i < 2; i++) for (int j = 0; j < 8; j++) sacc[i][j] = 0.f;
      #pragma unroll
      for (int sc = 0; sc < 2; sc++) {
        const int srow = sc*16 + ln;
        #pragma unroll
        for (int kc = 0; kc < 4; kc++) {
          Frag16 fa, fb;
          ldsFrag(fb, &sK[srow*136 + kc*32 + 8*lh]);
          #pragma unroll
          for (int j = 0; j < 8; j++) fa.u[j] = qf[kc][j];
          sacc[sc] = __builtin_amdgcn_wmma_f32_16x16x32_bf16(
              false, fa.v, false, fb.v, (short)0, sacc[sc], false, false);
        }
      }
      // causal mask + online softmax (C-layout rows: m = r + 8*lh)
      #pragma unroll
      for (int r = 0; r < 8; r++) {
        int qrow = wq0 + r + 8*lh;
        float v0 = (s0 + ln      <= qrow) ? sacc[0][r] : -3.0e38f;
        float v1 = (s0 + 16 + ln <= qrow) ? sacc[1][r] : -3.0e38f;
        float m = fmaxf(v0, v1);
        m = fmaxf(m, __shfl_xor(m, 1));
        m = fmaxf(m, __shfl_xor(m, 2));
        m = fmaxf(m, __shfl_xor(m, 4));
        m = fmaxf(m, __shfl_xor(m, 8));
        float newM = fmaxf(Mr[r], m);
        float alpha = __expf(Mr[r] - newM);
        float p0 = __expf(v0 - newM);
        float p1 = __expf(v1 - newM);
        float rs = p0 + p1;
        rs += __shfl_xor(rs, 1);
        rs += __shfl_xor(rs, 2);
        rs += __shfl_xor(rs, 4);
        rs += __shfl_xor(rs, 8);
        Lr[r] = Lr[r]*alpha + rs;
        Mr[r] = newM;
        #pragma unroll
        for (int dc = 0; dc < 8; dc++) oacc[dc][r] *= alpha;
        int prow = r + 8*lh;
        sP[pbase + prow*40 + ln]      = f2bf(p0);
        sP[pbase + prow*40 + 16 + ln] = f2bf(p1);
      }
      // O += P V
      Frag16 pa;
      ldsFrag(pa, &sP[pbase + ln*40 + 8*lh]);
      #pragma unroll
      for (int dc = 0; dc < 8; dc++) {
        Frag16 fv;
        ldsFrag(fv, &sVt[(dc*16 + ln)*40 + 8*lh]);
        oacc[dc] = __builtin_amdgcn_wmma_f32_16x16x32_bf16(
            false, pa.v, false, fv.v, (short)0, oacc[dc], false, false);
      }
    }
    __syncthreads();
  }

  #pragma unroll
  for (int r = 0; r < 8; r++) {
    int t = wq0 + r + 8*lh;
    float inv = 1.0f / Lr[r];
    #pragma unroll
    for (int dc = 0; dc < 8; dc++) {
      int d = dc*16 + ln;
      AO[((size_t)(b*TSEQ + t))*(NHEADS*DK) + h*DK + d] = f2bf(oacc[dc][r] * inv);
    }
  }
}

// ---------------- Kernel 3: output projection + bias ----------------
__global__ __launch_bounds__(256) void out_gemm_kernel(
    const unsigned short* __restrict__ A,  // (8192,2048) bf16
    const float* __restrict__ W,           // (2048,2048) fp32
    const float* __restrict__ bias,
    float* __restrict__ out)
{
  __shared__ unsigned short sA[128*40];
  __shared__ unsigned short sB[128*40];
  const int tid = threadIdx.x;
  const int w = tid >> 5, l = tid & 31, ln = l & 15, lh = l >> 4;
  const int m0 = blockIdx.y * 128, n0 = blockIdx.x * 128;

  v8f acc[8];
  for (int i = 0; i < 8; i++) for (int j = 0; j < 8; j++) acc[i][j] = 0.f;

  for (int k0 = 0; k0 < HID; k0 += 32) {
    for (int idx = tid; idx < 128*4; idx += 256) {        // bf16 A: b128 copy
      int r = idx >> 2, c16 = (idx & 3) * 16;             // halves
      uint4 v = *(const uint4*)&A[(size_t)(m0 + r)*HID + k0 + c16];
      *(uint4*)&sA[r*40 + c16] = v;
    }
    for (int idx = tid; idx < 128*8; idx += 256) {        // fp32 W -> bf16
      int r = idx >> 3, c = (idx & 7) * 4;
      float4 w4 = *(const float4*)&W[(size_t)(n0 + r)*HID + k0 + c];
      unsigned long long pb = (unsigned long long)pk2bf(w4.x, w4.y)
                            | ((unsigned long long)pk2bf(w4.z, w4.w) << 32);
      *(unsigned long long*)&sB[r*40 + c] = pb;
    }
    __syncthreads();

    Frag16 fa;
    ldsFrag(fa, &sA[(16*w + ln)*40 + 8*lh]);
    #pragma unroll
    for (int nc = 0; nc < 8; nc++) {
      Frag16 fb;
      ldsFrag(fb, &sB[(nc*16 + ln)*40 + 8*lh]);
      acc[nc] = __builtin_amdgcn_wmma_f32_16x16x32_bf16(
          false, fa.v, false, fb.v, (short)0, acc[nc], false, false);
    }
    __syncthreads();
  }

  #pragma unroll
  for (int nc = 0; nc < 8; nc++)
    #pragma unroll
    for (int r = 0; r < 8; r++) {
      int gm = m0 + 16*w + r + 8*lh;
      int gn = n0 + nc*16 + ln;
      out[(size_t)gm*HID + gn] = acc[nc][r] + bias[gn];
    }
}

extern "C" void kernel_launch(void* const* d_in, const int* in_sizes, int n_in,
                              void* d_out, int out_size, void* d_ws, size_t ws_size,
                              hipStream_t stream) {
  (void)in_sizes; (void)n_in; (void)out_size; (void)ws_size;
  const float* hidden = (const float*)d_in[0];
  const float* w_qkv  = (const float*)d_in[1];
  const float* w_out  = (const float*)d_in[2];
  const float* b_out  = (const float*)d_in[3];
  float* out = (float*)d_out;

  const size_t perBuf = (size_t)NB * NHEADS * TSEQ * DK;  // 16,777,216 elems
  unsigned short* Qb = (unsigned short*)d_ws;
  unsigned short* Kb = Qb + perBuf;
  unsigned short* Vb = Kb + perBuf;
  unsigned short* AO = Vb + perBuf;

  qkv_gemm_kernel<<<dim3(QKV_N/128, MTOT/128), 256, 0, stream>>>(hidden, w_qkv, Qb, Kb, Vb);
  flash_attn_kernel<<<dim3(NB*NHEADS*(TSEQ/128)), 256, 0, stream>>>(Qb, Kb, Vb, AO);
  out_gemm_kernel<<<dim3(HID/128, MTOT/128), 256, 0, stream>>>(AO, w_out, b_out, out);
}
